// OptimizedSGR_64613488001025
// MI455X (gfx1250) — compile-verified, matrix operand order checked by reading, not run on hardware
//
#include <hip/hip_runtime.h>

// ---------------------------------------------------------------------------
// Types for CDNA5 WMMA (wave32, v_wmma_f32_16x16x32_bf16)
// ---------------------------------------------------------------------------
typedef __bf16 bf16_t;
typedef bf16_t v16bf __attribute__((ext_vector_type(16)));
typedef float  v8f   __attribute__((ext_vector_type(8)));

#define WMMA_BF16(a, b, c) \
    __builtin_amdgcn_wmma_f32_16x16x32_bf16(false, (a), false, (b), (short)0, (c), false, false)

// Problem constants (match reference)
constexpr int kB  = 4;
constexpr int kT  = 8192;
constexpr int kD  = 512;
constexpr int kV  = 256;
constexpr int kM  = kB * kT;       // 32768 token rows
constexpr int kH  = 4 * kD;        // 2048 FFN hidden
constexpr int kNC = 128;           // scan chunks
constexpr int kTC = kT / kNC;      // 64 steps per chunk

// ---------------------------------------------------------------------------
// WMMA fragment load: row-major matrix with leading dim = K (A: [M][K],
// Bt: [N][K] pre-transposed). 16-bit A/B layout (ISA 7.12.2):
//   lane&15 -> M (or N) row; lane>>4 selects K-halves {h*8..h*8+7, 16+h*8..}
// ---------------------------------------------------------------------------
__device__ __forceinline__ v16bf ldfrag(const bf16_t* __restrict__ p, int hl) {
    union { v16bf v; float4 f[2]; } u;
    u.f[0] = *reinterpret_cast<const float4*>(p + hl * 8);
    u.f[1] = *reinterpret_cast<const float4*>(p + 16 + hl * 8);
    return u.v;
}

// Computes a 32x64 accumulator tile for this wave (2 M-tiles x 4 N-tiles).
// A:  [M][K] bf16 row-major, Bt: [N][K] bf16 row-major (i.e. B transposed).
__device__ __forceinline__ void gemm_acc(const bf16_t* __restrict__ A,
                                         const bf16_t* __restrict__ Bt,
                                         int K, int rowBase, int colBase,
                                         v8f acc[2][4]) {
    const int lane = threadIdx.x & 31;
    const int hl   = lane >> 4;
    const int r    = lane & 15;
    const bf16_t* ap[2];
    const bf16_t* bp[4];
#pragma unroll
    for (int i = 0; i < 2; ++i) ap[i] = A  + (size_t)(rowBase + i * 16 + r) * K;
#pragma unroll
    for (int j = 0; j < 4; ++j) bp[j] = Bt + (size_t)(colBase + j * 16 + r) * K;
    for (int k0 = 0; k0 < K; k0 += 32) {
        v16bf a[2], b[4];
#pragma unroll
        for (int i = 0; i < 2; ++i) a[i] = ldfrag(ap[i] + k0, hl);
#pragma unroll
        for (int j = 0; j < 4; ++j) b[j] = ldfrag(bp[j] + k0, hl);
#pragma unroll
        for (int i = 0; i < 2; ++i)
#pragma unroll
            for (int j = 0; j < 4; ++j)
                acc[i][j] = WMMA_BF16(a[i], b[j], acc[i][j]);
    }
}

// C/D layout (ISA 7.12.2): VGPR rr, lanes0-15 -> M=rr, lanes16-31 -> M=8+rr,
// N = lane&15. Epilogue macro iterates all 32x64 elements of this wave.
#define EPILOGUE_BEGIN()                                              \
    const int lane_e = threadIdx.x & 31;                              \
    const int hl_e   = lane_e >> 4;                                   \
    const int cl_e   = lane_e & 15;                                   \
    _Pragma("unroll") for (int mt = 0; mt < 2; ++mt)                  \
    _Pragma("unroll") for (int nt = 0; nt < 4; ++nt)                  \
    _Pragma("unroll") for (int rr = 0; rr < 8; ++rr) {                \
        const int row = rowBase + mt * 16 + hl_e * 8 + rr;            \
        const int col = colBase + nt * 16 + cl_e;                     \
        float v = acc[mt][nt][rr];
#define EPILOGUE_END() }

// Block tile 128x128: 8 waves = 4 (M) x 2 (N), each wave 32x64.
#define GEMM_PROLOGUE(KVAL)                                           \
    const int wid     = threadIdx.x >> 5;                             \
    const int rowBase = blockIdx.x * 128 + (wid & 3) * 32;            \
    const int colBase = blockIdx.y * 128 + (wid >> 2) * 64;           \
    v8f acc[2][4] = {};                                               \
    gemm_acc(A, Bt, (KVAL), rowBase, colBase, acc);

// ---------------------------------------------------------------------------
// GEMM kernels with fused epilogues
// ---------------------------------------------------------------------------

// alpha pre-act -> la = log(sigmoid+eps), bx = beta_scale*(1-alpha)*xn
__global__ __launch_bounds__(256) void k_gemm_alpha(
    const bf16_t* __restrict__ A, const bf16_t* __restrict__ Bt,
    const float* __restrict__ bfv, const float* __restrict__ alpha_bias,
    const float* __restrict__ beta_scale, const bf16_t* __restrict__ xn,
    float* __restrict__ la, float* __restrict__ bx) {
    GEMM_PROLOGUE(kD)
    const float ab = alpha_bias[0];
    const float bs = beta_scale[0];
    EPILOGUE_BEGIN()
        v += bfv[col] + ab;
        float alpha = 1.0f / (1.0f + __expf(-v));
        size_t idx = (size_t)row * kD + col;
        la[idx] = __logf(alpha + 1e-8f);
        bx[idx] = bs * (1.0f - alpha) * (float)xn[idx];
    EPILOGUE_END()
}

// gate: h = tanh(acc + br)*0.05 + h   (in-place residual)
__global__ __launch_bounds__(256) void k_gemm_gate(
    const bf16_t* __restrict__ A, const bf16_t* __restrict__ Bt,
    const float* __restrict__ brv, float* __restrict__ h) {
    GEMM_PROLOGUE(kD)
    EPILOGUE_BEGIN()
        float g = tanhf(v + brv[col]);
        size_t idx = (size_t)row * kD + col;
        h[idx] = g * 0.05f + h[idx];
    EPILOGUE_END()
}

// FFN up: act = gelu(acc + b1) as bf16, out ld = 2048
__global__ __launch_bounds__(256) void k_gemm_ffn1(
    const bf16_t* __restrict__ A, const bf16_t* __restrict__ Bt,
    const float* __restrict__ b1, bf16_t* __restrict__ act) {
    GEMM_PROLOGUE(kD)
    EPILOGUE_BEGIN()
        v += b1[col];
        float g = 0.5f * v * (1.0f + erff(v * 0.70710678118f));
        act[(size_t)row * kH + col] = (bf16_t)g;
    EPILOGUE_END()
}

// FFN down: h += acc + b2  (K = 2048)
__global__ __launch_bounds__(256) void k_gemm_ffn2(
    const bf16_t* __restrict__ A, const bf16_t* __restrict__ Bt,
    const float* __restrict__ b2, float* __restrict__ h) {
    GEMM_PROLOGUE(kH)
    EPILOGUE_BEGIN()
        size_t idx = (size_t)row * kD + col;
        h[idx] = h[idx] + v + b2[col];
    EPILOGUE_END()
}

// tied head: out = ho @ embed^T  (Bt = embed in bf16, already [V][D])
__global__ __launch_bounds__(256) void k_gemm_head(
    const bf16_t* __restrict__ A, const bf16_t* __restrict__ Bt,
    float* __restrict__ out) {
    GEMM_PROLOGUE(kD)
    EPILOGUE_BEGIN()
        out[(size_t)row * kV + col] = v;
    EPILOGUE_END()
}

// ---------------------------------------------------------------------------
// Elementwise / scan kernels
// ---------------------------------------------------------------------------

// embedding gather: one block per token
__global__ __launch_bounds__(256) void k_embed(
    const float* __restrict__ embed, const int* __restrict__ x,
    float* __restrict__ h) {
    const int tok = blockIdx.x;
    const int id  = x[tok];
    const int t   = threadIdx.x;
    h[(size_t)tok * kD + t]       = embed[(size_t)id * kD + t];
    h[(size_t)tok * kD + t + 256] = embed[(size_t)id * kD + t + 256];
}

// layernorm over D=512, one wave per row, output bf16
__global__ __launch_bounds__(256) void k_ln(
    const float* __restrict__ xin, const float* __restrict__ g,
    const float* __restrict__ b, bf16_t* __restrict__ out) {
    const int row  = blockIdx.x * 8 + (threadIdx.x >> 5);
    const int lane = threadIdx.x & 31;
    const float* xr = xin + (size_t)row * kD;
    float vals[16];
    float s = 0.0f, s2 = 0.0f;
#pragma unroll
    for (int i = 0; i < 16; ++i) {
        float v = xr[lane + i * 32];
        vals[i] = v;
        s += v; s2 += v * v;
    }
#pragma unroll
    for (int off = 16; off > 0; off >>= 1) {
        s  += __shfl_xor(s,  off, 32);
        s2 += __shfl_xor(s2, off, 32);
    }
    const float mean = s * (1.0f / kD);
    const float var  = s2 * (1.0f / kD) - mean * mean;
    const float rstd = rsqrtf(var + 1e-5f);
    bf16_t* orow = out + (size_t)row * kD;
#pragma unroll
    for (int i = 0; i < 16; ++i) {
        int col = lane + i * 32;
        orow[col] = (bf16_t)((vals[i] - mean) * rstd * g[col] + b[col]);
    }
}

// weight convert+transpose: in [K][N] f32 -> out [N][K] bf16
__global__ __launch_bounds__(256) void k_cvtT(
    const float* __restrict__ in, bf16_t* __restrict__ out, int K, int N) {
    int idx = blockIdx.x * 256 + threadIdx.x;
    if (idx < K * N) {
        int k = idx / N, n = idx % N;
        out[(size_t)n * K + k] = (bf16_t)in[idx];
    }
}

// plain f32 -> bf16 convert (embed table for head)
__global__ __launch_bounds__(256) void k_cvt(
    const float* __restrict__ in, bf16_t* __restrict__ out, int n) {
    int idx = blockIdx.x * 256 + threadIdx.x;
    if (idx < n) out[idx] = (bf16_t)in[idx];
}

// scan phase 1: per (b,chunk,d): local sum of la and local max of inclusive lac
__global__ __launch_bounds__(256) void k_scan1(
    const float* __restrict__ la, float* __restrict__ cs_sum,
    float* __restrict__ cs_max) {
    const int bc = blockIdx.x;           // b*kNC + c
    const int b  = bc / kNC, c = bc % kNC;
    const int d  = blockIdx.y * 256 + threadIdx.x;
    const int t0 = c * kTC;
    float s = 0.0f, m = -3.4e38f;
    for (int t = 0; t < kTC; ++t) {
        s += la[((size_t)(b * kT + t0 + t)) * kD + d];
        m = fmaxf(m, s);
    }
    size_t o = (size_t)bc * kD + d;
    cs_sum[o] = s;
    cs_max[o] = m;
}

// scan phase 2: per (b,d): exclusive scan of chunk sums + global max m
__global__ __launch_bounds__(256) void k_scan2(
    const float* __restrict__ cs_sum, const float* __restrict__ cs_max,
    float* __restrict__ cs_off, float* __restrict__ mbuf) {
    const int idx = blockIdx.x * 256 + threadIdx.x;   // b*kD + d
    const int b = idx >> 9, d = idx & 511;
    float run = 0.0f, m = -3.4e38f;
    for (int c = 0; c < kNC; ++c) {
        size_t o = ((size_t)(b * kNC + c)) * kD + d;
        cs_off[o] = run;
        m = fmaxf(m, run + cs_max[o]);
        run += cs_sum[o];
    }
    mbuf[idx] = m;
}

// scan phase 3: per chunk: partial sum of bx * exp(off + lac_local - m)
__global__ __launch_bounds__(256) void k_scan3(
    const float* __restrict__ la, const float* __restrict__ bx,
    const float* __restrict__ cs_off, const float* __restrict__ mbuf,
    float* __restrict__ cs_ps) {
    const int bc = blockIdx.x;
    const int b  = bc / kNC, c = bc % kNC;
    const int d  = blockIdx.y * 256 + threadIdx.x;
    const int t0 = c * kTC;
    const float base = cs_off[(size_t)bc * kD + d] - mbuf[b * kD + d];
    float lac = 0.0f, ps = 0.0f;
    for (int t = 0; t < kTC; ++t) {
        size_t idx = ((size_t)(b * kT + t0 + t)) * kD + d;
        lac += la[idx];
        ps += bx[idx] * __expf(base + lac);
    }
    cs_ps[(size_t)bc * kD + d] = ps;
}

// scan phase 4: per (b,d): exclusive scan of weighted partial sums
__global__ __launch_bounds__(256) void k_scan4(
    const float* __restrict__ cs_ps, float* __restrict__ cs_soff) {
    const int idx = blockIdx.x * 256 + threadIdx.x;
    const int b = idx >> 9, d = idx & 511;
    float run = 0.0f;
    for (int c = 0; c < kNC; ++c) {
        size_t o = ((size_t)(b * kNC + c)) * kD + d;
        cs_soff[o] = run;
        run += cs_ps[o];
    }
}

// scan phase 5: emit output = s*/(e+eps) as bf16 (next GEMM's A matrix)
__global__ __launch_bounds__(256) void k_scan5(
    const float* __restrict__ la, const float* __restrict__ bx,
    const float* __restrict__ cs_off, const float* __restrict__ mbuf,
    const float* __restrict__ cs_soff, bf16_t* __restrict__ out) {
    const int bc = blockIdx.x;
    const int b  = bc / kNC, c = bc % kNC;
    const int d  = blockIdx.y * 256 + threadIdx.x;
    const int t0 = c * kTC;
    const float base = cs_off[(size_t)bc * kD + d] - mbuf[b * kD + d];
    float lac = 0.0f, s = cs_soff[(size_t)bc * kD + d];
    for (int t = 0; t < kTC; ++t) {
        size_t idx = ((size_t)(b * kT + t0 + t)) * kD + d;
        lac += la[idx];
        float e = __expf(base + lac);
        s += bx[idx] * e;
        out[idx] = (bf16_t)(s / (e + 1e-8f));
    }
}

// ---------------------------------------------------------------------------
// Launch
// ---------------------------------------------------------------------------
extern "C" void kernel_launch(void* const* d_in, const int* in_sizes, int n_in,
                              void* d_out, int out_size, void* d_ws, size_t ws_size,
                              hipStream_t stream) {
    (void)in_sizes; (void)n_in; (void)out_size; (void)ws_size;
    auto F = [&](int i) { return (const float*)d_in[i]; };
    const float* embed = F(0);
    const int*   x     = (const int*)d_in[33];

    // workspace layout
    char* ws = (char*)d_ws;
    const size_t BTD = (size_t)kM * kD;
    size_t off = 0;
    float*  h   = (float*)(ws + off);  off += BTD * 4;      // 64 MB
    bf16_t* xnb = (bf16_t*)(ws + off); off += BTD * 2;      // 32 MB (xn/hn/ho)
    size_t offLA = off;
    float*  la  = (float*)(ws + off);  off += BTD * 4;      // 64 MB
    float*  bx  = (float*)(ws + off);  off += BTD * 4;      // 64 MB
    bf16_t* scb = (bf16_t*)(ws + off); off += BTD * 2;      // 32 MB scan out
    bf16_t* act = (bf16_t*)(ws + offLA);                    // 128 MB, reuses la+bx
    const size_t chunkN = (size_t)kB * kNC * kD;
    float* cs_sum  = (float*)(ws + off); off += chunkN * 4;
    float* cs_max  = (float*)(ws + off); off += chunkN * 4;
    float* cs_off  = (float*)(ws + off); off += chunkN * 4;
    float* cs_ps   = (float*)(ws + off); off += chunkN * 4;
    float* cs_soff = (float*)(ws + off); off += chunkN * 4;
    float* mbuf    = (float*)(ws + off); off += (size_t)kB * kD * 4;
    bf16_t* wfT[3], *wrT[3];
    for (int l = 0; l < 3; ++l) { wfT[l] = (bf16_t*)(ws + off); off += (size_t)kD * kD * 2; }
    for (int l = 0; l < 3; ++l) { wrT[l] = (bf16_t*)(ws + off); off += (size_t)kD * kD * 2; }
    bf16_t* w1T  = (bf16_t*)(ws + off); off += (size_t)kH * kD * 2;  // [2048][512]
    bf16_t* w2T  = (bf16_t*)(ws + off); off += (size_t)kD * kH * 2;  // [512][2048]
    bf16_t* embT = (bf16_t*)(ws + off); off += (size_t)kV * kD * 2;  // [256][512]

    // ---- weight conversion (bf16, pre-transposed to [N][K]) ----
    for (int l = 0; l < 3; ++l) {
        int base = 1 + l * 8;
        k_cvtT<<<(kD * kD + 255) / 256, 256, 0, stream>>>(F(base + 2), wfT[l], kD, kD);
        k_cvtT<<<(kD * kD + 255) / 256, 256, 0, stream>>>(F(base + 4), wrT[l], kD, kD);
    }
    k_cvtT<<<(kD * kH + 255) / 256, 256, 0, stream>>>(F(27), w1T, kD, kH);
    k_cvtT<<<(kH * kD + 255) / 256, 256, 0, stream>>>(F(29), w2T, kH, kD);
    k_cvt<<<(kV * kD + 255) / 256, 256, 0, stream>>>(embed, embT, kV * kD);

    // ---- embedding gather ----
    k_embed<<<kM, 256, 0, stream>>>(embed, x, h);

    // ---- 3 SGR layers ----
    const dim3 gD(kM / 128, kD / 128);      // (256, 4)
    const dim3 gScan(kB * kNC, 2);          // (512, 2)
    for (int l = 0; l < 3; ++l) {
        int base = 1 + l * 8;
        k_ln<<<kM / 8, 256, 0, stream>>>(h, F(base), F(base + 1), xnb);
        k_gemm_alpha<<<gD, 256, 0, stream>>>(xnb, wfT[l], F(base + 3),
                                             F(base + 6), F(base + 7), xnb, la, bx);
        k_scan1<<<gScan, 256, 0, stream>>>(la, cs_sum, cs_max);
        k_scan2<<<(kB * kD) / 256, 256, 0, stream>>>(cs_sum, cs_max, cs_off, mbuf);
        k_scan3<<<gScan, 256, 0, stream>>>(la, bx, cs_off, mbuf, cs_ps);
        k_scan4<<<(kB * kD) / 256, 256, 0, stream>>>(cs_ps, cs_soff);
        k_scan5<<<gScan, 256, 0, stream>>>(la, bx, cs_off, mbuf, cs_soff, scb);
        k_gemm_gate<<<gD, 256, 0, stream>>>(scb, wrT[l], F(base + 5), h);
    }

    // ---- FFN ----
    k_ln<<<kM / 8, 256, 0, stream>>>(h, F(25), F(26), xnb);
    k_gemm_ffn1<<<dim3(kM / 128, kH / 128), 256, 0, stream>>>(xnb, w1T, F(28), act);
    k_gemm_ffn2<<<gD, 256, 0, stream>>>(act, w2T, F(30), h);

    // ---- final LN + tied head ----
    k_ln<<<kM / 8, 256, 0, stream>>>(h, F(31), F(32), xnb);
    k_gemm_head<<<dim3(kM / 128, kV / 128), 256, 0, stream>>>(xnb, embT, (float*)d_out);
}
